// SelfAttention2d_89910845375320
// MI455X (gfx1250) — matrix operand order repaired and checked by reading, not compiled, not run on hardware
//
#include <hip/hip_runtime.h>

#define DEVINL __device__ __forceinline__

typedef _Float16 h16;
typedef __attribute__((ext_vector_type(8)))  _Float16 v8h;
typedef __attribute__((ext_vector_type(16))) _Float16 v16h;
typedef __attribute__((ext_vector_type(8)))  float    v8f;

constexpr int Cc = 256;       // channels
constexpr int Nn = 4096;      // tokens per batch (64*64)
constexpr int Bb = 4;         // batch
constexpr int NT = Bb * Nn;   // total tokens

DEVINL v16h cat16(v8h lo, v8h hi) {
  return __builtin_shufflevector(lo, hi, 0,1,2,3,4,5,6,7,8,9,10,11,12,13,14,15);
}
DEVINL v8h ldg8(const h16* p) { return *(const v8h*)p; }

DEVINL v8f wmma16(v16h a, v16h b, v8f c) {
  // D = A(16x32 f16) x B(32x16 f16) + C(16x16 f32)
  return __builtin_amdgcn_wmma_f32_16x16x32_f16(false, a, false, b, (short)0, c,
                                                false, false);
}

DEVINL float rowmax16(float v) {
  v = fmaxf(v, __shfl_xor(v, 1, 32));
  v = fmaxf(v, __shfl_xor(v, 2, 32));
  v = fmaxf(v, __shfl_xor(v, 4, 32));
  v = fmaxf(v, __shfl_xor(v, 8, 32));
  return v;
}
DEVINL float rowsum16(float v) {
  v += __shfl_xor(v, 1, 32);
  v += __shfl_xor(v, 2, 32);
  v += __shfl_xor(v, 4, 32);
  v += __shfl_xor(v, 8, 32);
  return v;
}

// ---------------- weight / input conversion ----------------

__global__ void cvt_w_kernel(const float* __restrict__ wq, const float* __restrict__ wk,
                             const float* __restrict__ wv, const float* __restrict__ wp,
                             h16* __restrict__ w16) {
  int i = blockIdx.x * blockDim.x + threadIdx.x;   // 0 .. 4*65536
  int mat = i >> 16, e = i & 65535;
  const float* srcs[4] = {wq, wk, wv, wp};
  w16[i] = (h16)srcs[mat][e];
  (void)mat; (void)e;
}

// xt16[b][n][c] = x[b][c][n]   (token-major f16)
__global__ void cvt_x_kernel(const float* __restrict__ x, h16* __restrict__ xt) {
  long i = (long)blockIdx.x * blockDim.x + threadIdx.x;   // NT*Cc threads
  int c = (int)(i & (Cc - 1));
  long nb = i >> 8;
  int n = (int)(nb & (Nn - 1));
  int b = (int)(nb >> 12);
  xt[i] = (h16)x[((long)b * Cc + c) * Nn + n];
}

// ---------------- QKV projection ----------------
// one wave = 16 tokens x full 256 output channels for q,k,v

__global__ __launch_bounds__(128) void qkv_kernel(
    const h16* __restrict__ xt,
    const h16* __restrict__ wq, const float* __restrict__ bq,
    const h16* __restrict__ wk, const float* __restrict__ bk,
    const h16* __restrict__ wv, const float* __restrict__ bv,
    h16* __restrict__ q, h16* __restrict__ k, h16* __restrict__ vT) {
  const int lane = threadIdx.x & 31, wid = threadIdx.x >> 5;
  const int t  = blockIdx.x * 4 + wid;
  const int b  = t >> 8;               // Nn/16 = 256 tiles per batch
  const int n0 = (t & 255) << 4;       // first token of tile
  const int m  = lane & 15, h = lane >> 4;
  const long base = ((long)b * Nn + n0) * Cc;

  // A fragments: 16 tokens x 256 channels
  v16h A[8];
#pragma unroll
  for (int kc = 0; kc < 8; ++kc) {
    const h16* p = xt + base + (long)m * Cc + kc * 32;
    A[kc] = cat16(ldg8(p + 8 * h), ldg8(p + 16 + 8 * h));
  }

  const h16* W[3]  = {wq, wk, wv};
  const float* Bi[3] = {bq, bk, bv};
  h16* O[2] = {q, k};

#pragma unroll
  for (int mat = 0; mat < 3; ++mat) {
    for (int dt = 0; dt < 16; ++dt) {
      const int d0 = dt * 16;
      v8f acc = {};
      const float bias = Bi[mat][d0 + m];   // column = lane&15
#pragma unroll
      for (int kc = 0; kc < 8; ++kc) {
        const h16* p = W[mat] + (long)(d0 + m) * Cc + kc * 32 + 8 * h;
        v16h Bf = cat16(ldg8(p), ldg8(p + 16));
        acc = wmma16(A[kc], Bf, acc);
      }
      if (mat < 2) {
        h16* dst = O[mat] + base;
#pragma unroll
        for (int r = 0; r < 8; ++r)
          dst[(long)(r + 8 * h) * Cc + d0 + m] = (h16)(acc[r] + bias);
      } else {   // V stored transposed [B][C][N]
#pragma unroll
        for (int r = 0; r < 8; ++r)
          vT[((long)b * Cc + d0 + m) * Nn + n0 + r + 8 * h] = (h16)(acc[r] + bias);
      }
    }
  }
}

// ---------------- flash attention ----------------
// one wave = 16 queries; online softmax over all 4096 keys, 32 keys / iter

__global__ __launch_bounds__(128) void attn_kernel(
    const h16* __restrict__ q, const h16* __restrict__ kk,
    const h16* __restrict__ vT, const h16* __restrict__ xt,
    h16* __restrict__ o16) {
  __shared__ __align__(16) h16 Qs[4][16 * 256];
  __shared__ __align__(16) h16 Ps[4][16 * 32];

  const int lane = threadIdx.x & 31, wid = threadIdx.x >> 5;
  const int t  = blockIdx.x * 4 + wid;
  const int b  = t >> 8;
  const int q0 = (t & 255) << 4;
  const int m  = lane & 15, h = lane >> 4;

  // stage this wave's 16x256 Q tile into LDS
  {
    const int row = lane >> 1, cb = (lane & 1) * 128;
    const h16* src = q + ((long)b * Nn + q0 + row) * Cc + cb;
    h16* dst = &Qs[wid][row * 256 + cb];
#pragma unroll
    for (int i = 0; i < 16; ++i)
      *(v8h*)(dst + i * 8) = *(const v8h*)(src + i * 8);
  }

  v8f o[16];
  {
    v8f z = {};
#pragma unroll
    for (int dt = 0; dt < 16; ++dt) o[dt] = z;
  }
  float mrun[8], lrun[8];
#pragma unroll
  for (int r = 0; r < 8; ++r) { mrun[r] = -1e30f; lrun[r] = 0.f; }

  const float sc = 0.0625f * 1.44269504089f;   // 1/sqrt(C) * log2(e)
  const long kbase = (long)b * Nn * Cc;

  for (int j0 = 0; j0 < Nn; j0 += 32) {
    if (j0 + 32 < Nn) {
      __builtin_prefetch(kk + kbase + (long)(j0 + 32 + m) * Cc, 0, 1);
      __builtin_prefetch(vT + ((long)b * Cc + m) * Nn + j0 + 32, 0, 1);
    }
    // ---- S = Q * K^T for 32 keys (two 16x16 f32 tiles) ----
    v8f s0 = {}, s1 = {};
#pragma unroll
    for (int kc = 0; kc < 8; ++kc) {
      const h16* qa = &Qs[wid][m * 256 + kc * 32];
      v16h A = cat16(*(const v8h*)(qa + 8 * h), *(const v8h*)(qa + 16 + 8 * h));
      const h16* kp = kk + kbase + (long)(j0 + m) * Cc + kc * 32 + 8 * h;
      v16h B0 = cat16(ldg8(kp), ldg8(kp + 16));
      v16h B1 = cat16(ldg8(kp + 16 * Cc), ldg8(kp + 16 * Cc + 16));
      s0 = wmma16(A, B0, s0);
      s1 = wmma16(A, B1, s1);
    }
    // ---- online softmax update (rows live across 16 lanes) ----
    float alpha[8];
#pragma unroll
    for (int r = 0; r < 8; ++r) {
      float a0 = s0[r] * sc, a1 = s1[r] * sc;
      float mx = rowmax16(fmaxf(a0, a1));
      float mn = fmaxf(mrun[r], mx);
      alpha[r] = exp2f(mrun[r] - mn);
      mrun[r] = mn;
      float p0 = exp2f(a0 - mn);
      float p1 = exp2f(a1 - mn);
      lrun[r] = lrun[r] * alpha[r] + rowsum16(p0 + p1);
      Ps[wid][(r + 8 * h) * 32 + m]      = (h16)p0;   // C-layout -> LDS
      Ps[wid][(r + 8 * h) * 32 + 16 + m] = (h16)p1;
    }
#pragma unroll
    for (int dt = 0; dt < 16; ++dt)
#pragma unroll
      for (int r = 0; r < 8; ++r) o[dt][r] *= alpha[r];

    // ---- reload P in A-fragment layout, accumulate O += P*V ----
    const h16* pp = &Ps[wid][m * 32 + 8 * h];
    v16h P = cat16(*(const v8h*)pp, *(const v8h*)(pp + 16));
#pragma unroll
    for (int dt = 0; dt < 16; ++dt) {
      const h16* vp = vT + ((long)b * Cc + dt * 16 + m) * Nn + j0 + 8 * h;
      v16h Bv = cat16(ldg8(vp), ldg8(vp + 16));
      o[dt] = wmma16(P, Bv, o[dt]);
    }
  }

  // ---- normalize, add residual, store f16 [B][N][C] ----
  float rl[8];
#pragma unroll
  for (int r = 0; r < 8; ++r) rl[r] = 1.0f / lrun[r];
#pragma unroll
  for (int dt = 0; dt < 16; ++dt) {
#pragma unroll
    for (int r = 0; r < 8; ++r) {
      long idx = ((long)b * Nn + q0 + r + 8 * h) * Cc + dt * 16 + m;
      float val = o[dt][r] * rl[r] + (float)xt[idx];
      o16[idx] = (h16)val;
    }
  }
}

// ---------------- output projection ----------------

__global__ __launch_bounds__(128) void proj_kernel(
    const h16* __restrict__ o16, const h16* __restrict__ wp,
    const float* __restrict__ bp, float* __restrict__ out) {
  const int lane = threadIdx.x & 31, wid = threadIdx.x >> 5;
  const int t  = blockIdx.x * 4 + wid;
  const int b  = t >> 8;
  const int n0 = (t & 255) << 4;
  const int m  = lane & 15, h = lane >> 4;
  const long base = ((long)b * Nn + n0) * Cc;

  v16h A[8];
#pragma unroll
  for (int kc = 0; kc < 8; ++kc) {
    const h16* p = o16 + base + (long)m * Cc + kc * 32;
    A[kc] = cat16(ldg8(p + 8 * h), ldg8(p + 16 + 8 * h));
  }
  for (int dt = 0; dt < 16; ++dt) {
    const int d0 = dt * 16;
    v8f acc = {};
    const float bias = bp[d0 + m];
#pragma unroll
    for (int kc = 0; kc < 8; ++kc) {
      const h16* p = wp + (long)(d0 + m) * Cc + kc * 32 + 8 * h;
      v16h Bf = cat16(ldg8(p), ldg8(p + 16));
      acc = wmma16(A[kc], Bf, acc);
    }
#pragma unroll
    for (int r = 0; r < 8; ++r)
      out[((long)b * Cc + d0 + m) * Nn + n0 + r + 8 * h] = acc[r] + bias;
  }
}

// ---------------- host launch ----------------

extern "C" void kernel_launch(void* const* d_in, const int* in_sizes, int n_in,
                              void* d_out, int out_size, void* d_ws, size_t ws_size,
                              hipStream_t stream) {
  const float* x  = (const float*)d_in[0];
  const float* wq = (const float*)d_in[1];
  const float* bq = (const float*)d_in[2];
  const float* wk = (const float*)d_in[3];
  const float* bk = (const float*)d_in[4];
  const float* wv = (const float*)d_in[5];
  const float* bv = (const float*)d_in[6];
  const float* wp = (const float*)d_in[7];
  const float* bp = (const float*)d_in[8];

  // f16 workspace layout (halves): 4 weight mats, xt, q, k, vT, o  (~42.5 MB)
  h16* ws   = (h16*)d_ws;
  h16* w16  = ws;                              // 4 * 65536
  h16* xt   = ws + 4L * 65536;                 // NT*Cc
  h16* qbuf = xt + (long)NT * Cc;
  h16* kbuf = qbuf + (long)NT * Cc;
  h16* vT   = kbuf + (long)NT * Cc;
  h16* obuf = vT + (long)NT * Cc;

  cvt_w_kernel<<<(4 * 65536) / 256, 256, 0, stream>>>(wq, wk, wv, wp, w16);
  cvt_x_kernel<<<((long)NT * Cc) / 256, 256, 0, stream>>>(x, xt);

  const int blocks = NT / 64;   // 4 wave-tiles (16 tokens each) per block
  qkv_kernel<<<blocks, 128, 0, stream>>>(xt, w16, bq, w16 + 65536, bk,
                                         w16 + 2 * 65536, bv, qbuf, kbuf, vT);
  attn_kernel<<<blocks, 128, 0, stream>>>(qbuf, kbuf, vT, xt, obuf);
  proj_kernel<<<blocks, 128, 0, stream>>>(obuf, w16 + 3 * 65536, bp, (float*)d_out);

  (void)in_sizes; (void)n_in; (void)out_size; (void)ws_size;
}
